// M_GAT_48249662603675
// MI455X (gfx1250) — matrix operand order repaired and checked by reading, not compiled, not run on hardware
//
#include <hip/hip_runtime.h>
#include <hip/hip_bf16.h>
#include <stdint.h>

typedef __attribute__((ext_vector_type(16))) _Float16 v16h;
typedef __attribute__((ext_vector_type(8)))  float    v8f;

// ---------------------------------------------------------------- utilities
__global__ void fill_f32(float* p, float v, size_t n) {
    size_t t = (size_t)blockIdx.x * blockDim.x + threadIdx.x;
    if (t < n) p[t] = v;
}
__global__ void fill_u32(uint32_t* p, uint32_t v, size_t n) {
    size_t t = (size_t)blockIdx.x * blockDim.x + threadIdx.x;
    if (t < n) p[t] = v;
}

// ---------------------------------------------------------------- embedding concat -> padded f16 [N,64]
__global__ void embed_x16(const float* __restrict__ inputs,
                          const float* __restrict__ e0,
                          const float* __restrict__ e1,
                          const float* __restrict__ e2,
                          _Float16* __restrict__ x16, int N) {
    int t = blockIdx.x * blockDim.x + threadIdx.x;
    if (t >= N * 64) return;
    int n = t >> 6, c = t & 63;
    const float* in = inputs + (size_t)n * 33;
    float v = 0.f;
    if (c < 8) {
        int i0 = (int)in[0];
        v = e0[i0 * 8 + c];
    } else if (c < 11) {
        int i1 = (int)in[1];
        v = e1[i1 * 3 + (c - 8)];
    } else if (c < 17) {
        int i2 = (int)in[2];
        v = e2[i2 * 6 + (c - 11)];
    } else if (c < 47) {
        v = in[3 + (c - 17)];
    }
    x16[t] = (_Float16)v;
}

// ---------------------------------------------------------------- pack W (f32, [T, rs, KOUT]) into WMMA
// B-fragment order (f16): dst[et][kstep][j][lane][e], where each lane's 16 halfs are a contiguous
// 32B chunk = exactly one v16h B fragment for tile (kstep, j).
//   k = kstep*32 + 8*(lane/16) + (e%8) + 16*(e/8),  n = j*16 + (lane%16)
template <int KIN, int KOUT>
__global__ void pack_w(const float* __restrict__ src, _Float16* __restrict__ dst, int T, int rs) {
    constexpr int NT = KOUT / 16;
    int t = blockIdx.x * blockDim.x + threadIdx.x;
    int per = KIN * KOUT;
    if (t >= T * per) return;
    int et = t / per;
    int u = t % per;
    int fragidx = u >> 9;          // /512 halfs per fragment-group (32 lanes x 16)
    int within = u & 511;
    int lane = within >> 4;
    int e = within & 15;
    int j = fragidx % NT;
    int kb = fragidx / NT;
    int k = kb * 32 + 8 * (lane >> 4) + (e & 7) + ((e >> 3) << 4);
    int n = j * 16 + (lane & 15);
    float v = (k < rs) ? src[(size_t)et * rs * KOUT + (size_t)k * KOUT + n] : 0.f;
    dst[t] = (_Float16)v;
}

// ---------------------------------------------------------------- WMMA GEMM: C[N,KOUT] = A16[N,KIN] @ Wp
// A staged in LDS (coalesced), B fragments loaded directly from packed global W (cache-resident).
template <int KIN, int KOUT>
__global__ void gemm_wmma(const _Float16* __restrict__ A, const _Float16* __restrict__ Wp,
                          float* __restrict__ C, int N) {
    constexpr int NT  = KOUT / 16;
    constexpr int AST = 36;  // padded LDS row stride (halfs)
    __shared__ _Float16 As[128 * AST];

    const int tid  = threadIdx.x;
    const int w    = tid >> 5;
    const int lane = tid & 31;
    const int lrow = lane & 15;
    const int koff = (lane < 16) ? 0 : 8;
    const int rowbase = blockIdx.x * 128;

    v8f acc[NT];
#pragma unroll
    for (int j = 0; j < NT; j++)
#pragma unroll
        for (int k = 0; k < 8; k++) acc[j][k] = 0.f;

    for (int kb = 0; kb < KIN / 32; kb++) {
        const int k0 = kb * 32;
        __syncthreads();
        // --- A slab: 128 rows x 32 halfs = 2048 dwords, coalesced
#pragma unroll
        for (int i = 0; i < 8; i++) {
            int d = tid + i * 256;
            int r = d >> 4;
            int col = (d & 15) * 2;
            int gr = rowbase + r;
            if (gr >= N) gr = N - 1;
            uint32_t v = *reinterpret_cast<const uint32_t*>(A + (size_t)gr * KIN + k0 + col);
            *reinterpret_cast<uint32_t*>(&As[r * AST + col]) = v;
        }
        __syncthreads();

        // A fragment: lane L holds row (w*16 + L%16); halfs 0..7 = K(koff..+7), 8..15 = K(16+koff..+7)
        v16h a;
        const _Float16* ap = &As[(w * 16 + lrow) * AST];
#pragma unroll
        for (int i = 0; i < 8; i++) {
            a[i]     = ap[koff + i];
            a[8 + i] = ap[16 + koff + i];
        }

        // B fragments: one 32B contiguous load per lane per tile from packed W
        const _Float16* wbase = Wp + ((size_t)kb * NT * 32 + lane) * 16;
#pragma unroll
        for (int j = 0; j < NT; j++) {
            v16h b = *reinterpret_cast<const v16h*>(wbase + (size_t)j * 512);
            acc[j] = __builtin_amdgcn_wmma_f32_16x16x32_f16(
                false, a, false, b, (short)0, acc[j], false, false);
        }
    }

    // C layout: VGPR i -> row i (lanes 0-15) / row 8+i (lanes 16-31), col = lane%16
#pragma unroll
    for (int j = 0; j < NT; j++) {
#pragma unroll
        for (int i = 0; i < 8; i++) {
            int r = rowbase + w * 16 + ((lane < 16) ? i : 8 + i);
            if (r < N) C[(size_t)r * KOUT + j * 16 + lrow] = acc[j][i];
        }
    }
}

// ---------------------------------------------------------------- attention scores el/er
__global__ void scores_k(const float* __restrict__ feat, const float* __restrict__ al,
                         const float* __restrict__ ar, float* __restrict__ el,
                         float* __restrict__ er, int N, int H, int D) {
    int t = blockIdx.x * blockDim.x + threadIdx.x;
    if (t >= N * H) return;
    int n = t / H, h = t % H;
    const float* f = feat + (size_t)n * H * D + (size_t)h * D;
    const float* a = al + h * D;
    const float* b = ar + h * D;
    float sl = 0.f, sr = 0.f;
    for (int d = 0; d < D; d++) {
        float v = f[d];
        sl += v * a[d];
        sr += v * b[d];
    }
    el[t] = sl;
    er[t] = sr;
}

// monotonic float<->uint mapping for atomic max
__device__ __forceinline__ uint32_t f2ord(float f) {
    uint32_t u = __float_as_uint(f);
    return (u & 0x80000000u) ? ~u : (u | 0x80000000u);
}
__device__ __forceinline__ float ord2f(uint32_t k) {
    uint32_t u = (k & 0x80000000u) ? (k & 0x7FFFFFFFu) : ~k;
    return __uint_as_float(u);
}

// edge pass 1: e = lrelu(el[src]+er[dst], 0.2); segment max into mkey[dst]
__global__ void edge1_k(const float* __restrict__ el, const float* __restrict__ er,
                        const int* __restrict__ src, const int* __restrict__ dst,
                        float* __restrict__ ebuf, uint32_t* __restrict__ mkey,
                        int E, int H) {
    int t = blockIdx.x * blockDim.x + threadIdx.x;
    if (t >= E * H) return;
    int e = t / H, h = t % H;
    int s = src[e], d = dst[e];
    float v = el[s * H + h] + er[d * H + h];
    v = (v > 0.f) ? v : 0.2f * v;
    ebuf[t] = v;
    atomicMax(&mkey[d * H + h], f2ord(v));
}

// decode max keys in place; non-finite -> 0
__global__ void finm_k(uint32_t* __restrict__ m, int n) {
    int t = blockIdx.x * blockDim.x + threadIdx.x;
    if (t >= n) return;
    float v = ord2f(m[t]);
    uint32_t bits = __float_as_uint(v);
    if ((bits & 0x7F800000u) == 0x7F800000u) v = 0.f;  // inf/nan
    reinterpret_cast<float*>(m)[t] = v;
}

// edge pass 2: ex = exp(e - m[dst]); segment sum into sbuf[dst]
__global__ void edge2_k(const float* __restrict__ mf, const int* __restrict__ dst,
                        float* __restrict__ ebuf, float* __restrict__ sbuf,
                        int E, int H) {
    int t = blockIdx.x * blockDim.x + threadIdx.x;
    if (t >= E * H) return;
    int e = t / H, h = t % H;
    int d = dst[e];
    float ex = expf(ebuf[t] - mf[d * H + h]);
    ebuf[t] = ex;
    atomicAdd(&sbuf[d * H + h], ex);
}

// edge pass 3: out[dst] += alpha * feat[src], 4 floats per thread
__global__ void agg_k(const float* __restrict__ feat, const float* __restrict__ ebuf,
                      const float* __restrict__ sbuf, const int* __restrict__ src,
                      const int* __restrict__ dst, float* __restrict__ outb,
                      int E, int H, int D, int KOUT) {
    int qpe = KOUT / 4;
    long long t = (long long)blockIdx.x * blockDim.x + threadIdx.x;
    if (t >= (long long)E * qpe) return;
    int e = (int)(t / qpe);
    int q = (int)(t % qpe);
    int f0 = q * 4;
    int h = f0 / D;
    int s = src[e], d = dst[e];
    float alpha = ebuf[e * H + h] / fmaxf(sbuf[d * H + h], 1e-9f);
    const float4 fv = *reinterpret_cast<const float4*>(feat + (size_t)s * KOUT + f0);
    float* o = outb + (size_t)d * KOUT + f0;
    atomicAdd(o + 0, alpha * fv.x);
    atomicAdd(o + 1, alpha * fv.y);
    atomicAdd(o + 2, alpha * fv.z);
    atomicAdd(o + 3, alpha * fv.w);
}

// bias + leaky_relu(0.01), emit f16 activations for next layer's GEMM
__global__ void bias_act_k(const float* __restrict__ outb, const float* __restrict__ b,
                           _Float16* __restrict__ h16, int N, int K) {
    int t = blockIdx.x * blockDim.x + threadIdx.x;
    if (t >= N * K) return;
    int f = t % K;
    float v = outb[t] + b[f];
    v = (v > 0.f) ? v : 0.01f * v;
    h16[t] = (_Float16)v;
}

// final layer: bias, scale by 1/T, accumulate into d_out
__global__ void bias_accum_k(const float* __restrict__ outb, const float* __restrict__ b,
                             float* __restrict__ dout, int N, int K, float scale) {
    int t = blockIdx.x * blockDim.x + threadIdx.x;
    if (t >= N * K) return;
    int f = t % K;
    dout[t] += (outb[t] + b[f]) * scale;
}

// ---------------------------------------------------------------- host side
static inline unsigned gridFor(size_t n, int b) { return (unsigned)((n + b - 1) / b); }

extern "C" void kernel_launch(void* const* d_in, const int* in_sizes, int n_in,
                              void* d_out, int out_size, void* d_ws, size_t ws_size,
                              hipStream_t stream) {
    const float* inputs = (const float*)d_in[0];
    const float* emb0   = (const float*)d_in[1];
    const float* emb1   = (const float*)d_in[2];
    const float* emb2   = (const float*)d_in[3];
    const float* W1     = (const float*)d_in[4];
    const float* al1    = (const float*)d_in[5];
    const float* ar1    = (const float*)d_in[6];
    const float* b1     = (const float*)d_in[7];
    const float* W2     = (const float*)d_in[8];
    const float* al2    = (const float*)d_in[9];
    const float* ar2    = (const float*)d_in[10];
    const float* b2     = (const float*)d_in[11];
    const float* W3     = (const float*)d_in[12];
    const float* al3    = (const float*)d_in[13];
    const float* ar3    = (const float*)d_in[14];
    const float* b3     = (const float*)d_in[15];
    const int*   src    = (const int*)d_in[16];
    const int*   dst    = (const int*)d_in[17];
    float* dout = (float*)d_out;

    const int N = in_sizes[0] / 33;
    const int E = in_sizes[16] / 3;
    const int T = 3;

    // workspace bump allocator
    char* base = (char*)d_ws;
    size_t off = 0;
    auto alloc = [&](size_t bytes) -> char* {
        off = (off + 255) & ~(size_t)255;
        char* p = base + off;
        off += bytes;
        return p;
    };
    _Float16* W1h  = (_Float16*)alloc((size_t)T * 64 * 192 * 2);
    _Float16* W2h  = (_Float16*)alloc((size_t)T * 192 * 192 * 2);
    _Float16* W3h  = (_Float16*)alloc((size_t)T * 192 * 128 * 2);
    _Float16* x16  = (_Float16*)alloc((size_t)N * 64 * 2);
    _Float16* h16  = (_Float16*)alloc((size_t)N * 192 * 2);
    float*    feat = (float*)alloc((size_t)N * 192 * 4);
    float*    outb = (float*)alloc((size_t)N * 192 * 4);
    float*    el   = (float*)alloc((size_t)N * 3 * 4);
    float*    er   = (float*)alloc((size_t)N * 3 * 4);
    uint32_t* mkey = (uint32_t*)alloc((size_t)N * 3 * 4);
    float*    sbuf = (float*)alloc((size_t)N * 3 * 4);
    float*    ebuf = (float*)alloc((size_t)E * 3 * 4);
    (void)ws_size;

    const int B = 256;

    // one-time prep: zero output, build f16 inputs, pack weights into fragment order
    fill_f32<<<gridFor((size_t)N * 128, B), B, 0, stream>>>(dout, 0.f, (size_t)N * 128);
    embed_x16<<<gridFor((size_t)N * 64, B), B, 0, stream>>>(inputs, emb0, emb1, emb2, x16, N);
    pack_w<64, 192><<<gridFor((size_t)T * 64 * 192, B), B, 0, stream>>>(W1, W1h, T, 47);
    pack_w<192, 192><<<gridFor((size_t)T * 192 * 192, B), B, 0, stream>>>(W2, W2h, T, 192);
    pack_w<192, 128><<<gridFor((size_t)T * 192 * 128, B), B, 0, stream>>>(W3, W3h, T, 192);

    auto run_attn = [&](const float* al_, const float* ar_, const int* sp, const int* dp,
                        int H, int D, int KOUT) {
        scores_k<<<gridFor((size_t)N * H, B), B, 0, stream>>>(feat, al_, ar_, el, er, N, H, D);
        fill_u32<<<gridFor((size_t)N * H, B), B, 0, stream>>>(mkey, 0x007FFFFFu, (size_t)N * H);
        fill_f32<<<gridFor((size_t)N * H, B), B, 0, stream>>>(sbuf, 0.f, (size_t)N * H);
        edge1_k<<<gridFor((size_t)E * H, B), B, 0, stream>>>(el, er, sp, dp, ebuf, mkey, E, H);
        finm_k<<<gridFor((size_t)N * H, B), B, 0, stream>>>(mkey, N * H);
        edge2_k<<<gridFor((size_t)E * H, B), B, 0, stream>>>((const float*)mkey, dp, ebuf, sbuf, E, H);
        fill_f32<<<gridFor((size_t)N * KOUT, B), B, 0, stream>>>(outb, 0.f, (size_t)N * KOUT);
        agg_k<<<gridFor((size_t)E * (KOUT / 4), B), B, 0, stream>>>(feat, ebuf, sbuf, sp, dp, outb,
                                                                    E, H, D, KOUT);
    };

    const unsigned gGemm = gridFor((size_t)N, 128);

    for (int t = 0; t < T; t++) {
        const int* sp = src + (size_t)t * E;
        const int* dp = dst + (size_t)t * E;

        // ---- layer 1: [N,64(pad 47)] @ [64,192], H=3, D=64, act
        gemm_wmma<64, 192><<<gGemm, 256, 0, stream>>>(x16, W1h + (size_t)t * 64 * 192, feat, N);
        run_attn(al1 + (size_t)t * 3 * 64, ar1 + (size_t)t * 3 * 64, sp, dp, 3, 64, 192);
        bias_act_k<<<gridFor((size_t)N * 192, B), B, 0, stream>>>(outb, b1 + (size_t)t * 192, h16, N, 192);

        // ---- layer 2: [N,192] @ [192,192], H=3, D=64, act
        gemm_wmma<192, 192><<<gGemm, 256, 0, stream>>>(h16, W2h + (size_t)t * 192 * 192, feat, N);
        run_attn(al2 + (size_t)t * 3 * 64, ar2 + (size_t)t * 3 * 64, sp, dp, 3, 64, 192);
        bias_act_k<<<gridFor((size_t)N * 192, B), B, 0, stream>>>(outb, b2 + (size_t)t * 192, h16, N, 192);

        // ---- layer 3: [N,192] @ [192,128], H=1, D=128, no act; mean into d_out
        gemm_wmma<192, 128><<<gGemm, 256, 0, stream>>>(h16, W3h + (size_t)t * 192 * 128, feat, N);
        run_attn(al3 + (size_t)t * 128, ar3 + (size_t)t * 128, sp, dp, 1, 128, 128);
        bias_accum_k<<<gridFor((size_t)N * 128, B), B, 0, stream>>>(outb, b3 + (size_t)t * 128,
                                                                    dout, N, 128, 1.0f / 3.0f);
    }
}
